// TopKRouter_56478819942538
// MI455X (gfx1250) — matrix operand order repaired and checked by reading, not compiled, not run on hardware
//
#include <hip/hip_runtime.h>
#include <hip/hip_bf16.h>
#include <math.h>

#define TOKENS   16384
#define DIM      4096
#define NEXP     64
#define TOPK     2
#define BM       64            // tokens per block
#define KC       64            // K chunk staged in LDS
#define NCHUNK   (DIM / KC)    // 64
#define NWAVES   (BM / 16)     // 4 waves
#define NTHREADS (NWAVES * 32) // 128 threads
#define LDAF     68            // fp32 x-tile LDS row stride (elems): 272B = 17*16B
#define LDB      72            // bf16 w-tile LDS row stride (elems): 144B = 9*16B

typedef __attribute__((ext_vector_type(16))) __bf16 v16bf;
typedef __attribute__((ext_vector_type(8)))  __bf16 v8bf;
typedef __attribute__((ext_vector_type(8)))  float  v8f;
typedef unsigned int v4u __attribute__((ext_vector_type(4)));
typedef int          v8i __attribute__((ext_vector_type(8)));
typedef int          v4i __attribute__((ext_vector_type(4)));

#if __has_builtin(__builtin_amdgcn_tensor_load_to_lds) && \
    __has_builtin(__builtin_amdgcn_s_wait_tensorcnt)
#define USE_TDM 1
#else
#define USE_TDM 0
#endif

__device__ __forceinline__ unsigned short f2bf(float f) {
    unsigned u = __float_as_uint(f);
    return (unsigned short)((u + 0x7FFFu + ((u >> 16) & 1u)) >> 16); // RNE truncate
}

// Build A fragment: 16 fp32 from LDS (2x 32B runs) -> v16bf via native HW converts
__device__ __forceinline__ v16bf cvt_frag(const float* p0, const float* p1) {
    const float4 q0 = reinterpret_cast<const float4*>(p0)[0];
    const float4 q1 = reinterpret_cast<const float4*>(p0)[1];
    const float4 q2 = reinterpret_cast<const float4*>(p1)[0];
    const float4 q3 = reinterpret_cast<const float4*>(p1)[1];
    v16bf r;
    r[0]  = (__bf16)q0.x; r[1]  = (__bf16)q0.y; r[2]  = (__bf16)q0.z; r[3]  = (__bf16)q0.w;
    r[4]  = (__bf16)q1.x; r[5]  = (__bf16)q1.y; r[6]  = (__bf16)q1.z; r[7]  = (__bf16)q1.w;
    r[8]  = (__bf16)q2.x; r[9]  = (__bf16)q2.y; r[10] = (__bf16)q2.z; r[11] = (__bf16)q2.w;
    r[12] = (__bf16)q3.x; r[13] = (__bf16)q3.y; r[14] = (__bf16)q3.z; r[15] = (__bf16)q3.w;
    return r;
}

// B fragment: 16 bf16 from LDS (2x 16B runs)
__device__ __forceinline__ v16bf load_frag(const unsigned short* p0, const unsigned short* p1) {
    v16bf r;
    *reinterpret_cast<v8bf*>(&r)       = *reinterpret_cast<const v8bf*>(p0);
    *(reinterpret_cast<v8bf*>(&r) + 1) = *reinterpret_cast<const v8bf*>(p1);
    return r;
}

#if USE_TDM
// 2D tile Global->LDS via Tensor Data Mover (D# per CDNA5 ISA ch.8).
// pad_amount in dwords (0 = no padding); pad_interval encoded per spec (5 -> 64 dw, 4 -> 32 dw)
__device__ __forceinline__ void tdm_load_2d(unsigned lds_off, const void* gptr,
                                            unsigned data_size_log,
                                            unsigned tensor_d0, unsigned tensor_d1,
                                            unsigned tile_d0, unsigned tile_d1,
                                            unsigned long long stride0_elems,
                                            unsigned pad_interval, unsigned pad_amount_dw) {
    const unsigned long long ga = (unsigned long long)(uintptr_t)gptr;
    v4u g0;
    g0[0] = 1u;                                            // count=1 (valid), user mode
    g0[1] = lds_off;                                       // lds_addr (bytes)
    g0[2] = (unsigned)(ga & 0xFFFFFFFFu);                  // global_addr[31:0]
    g0[3] = (unsigned)((ga >> 32) & 0x01FFFFFFu) | (2u << 30); // global_addr[56:32] | type=2
    const unsigned pe = (pad_amount_dw != 0u) ? 1u : 0u;
    v8i g1;
    g1[0] = (int)((data_size_log << 16) | (pe << 20) | (pad_interval << 22) |
                  (pe ? ((pad_amount_dw - 1u) << 25) : 0u)); // wg_mask=0, no atomic barrier
    g1[1] = (int)((tensor_d0 & 0xFFFFu) << 16);            // tensor_dim0[15:0]
    g1[2] = (int)((tensor_d0 >> 16) | ((tensor_d1 & 0xFFFFu) << 16));
    g1[3] = (int)((tensor_d1 >> 16) | ((tile_d0 & 0xFFFFu) << 16));
    g1[4] = (int)(tile_d1 & 0xFFFFu);                      // tile_dim1; tile_dim2=0
    g1[5] = (int)(stride0_elems & 0xFFFFFFFFull);          // tensor_dim0_stride[31:0]
    g1[6] = (int)((stride0_elems >> 32) & 0xFFFFull);      // stride[47:32]; dim1_stride=0
    g1[7] = 0;
    const v4i gz = {0, 0, 0, 0};
#if defined(__clang_major__) && (__clang_major__ >= 23)
    const v8i gz8 = {0, 0, 0, 0, 0, 0, 0, 0};
    __builtin_amdgcn_tensor_load_to_lds(g0, g1, gz, gz, gz8, 0);
#else
    __builtin_amdgcn_tensor_load_to_lds(g0, g1, gz, gz, 0);
#endif
}
#endif // USE_TDM

// one-time fp32 -> bf16 conversion of w_gating into workspace
__global__ void convert_w(const float* __restrict__ w, unsigned short* __restrict__ wbf) {
    const int i = blockIdx.x * blockDim.x + threadIdx.x; // over NEXP*DIM/4
    if (i < NEXP * DIM / 4) {
        const float4 v = reinterpret_cast<const float4*>(w)[i];
        ushort4 o;
        o.x = f2bf(v.x); o.y = f2bf(v.y); o.z = f2bf(v.z); o.w = f2bf(v.w);
        reinterpret_cast<ushort4*>(wbf)[i] = o;
    }
}

// logits GEMM (bf16 WMMA, TDM-fed) + softmax + top-2 + scores + loss partials
__global__ __launch_bounds__(NTHREADS)
void router_main(const float* __restrict__ x, const unsigned short* __restrict__ wbf,
                 float* __restrict__ out, float* __restrict__ ws) {
#if USE_TDM
    __shared__ __align__(16) float          sXF[2][BM * LDAF];
    __shared__ __align__(16) unsigned short sWB[2][NEXP * LDB];
#else
    __shared__ __align__(16) float          sXF[1][BM * LDAF];
    __shared__ __align__(16) unsigned short sWB[1][NEXP * LDB];
#endif
    __shared__ float sMe[NEXP];
    __shared__ float sCe[NEXP];
    __shared__ float sZ;

    const int tid      = threadIdx.x;
    const int wave     = tid >> 5;
    const int lane     = tid & 31;
    const int laneLow  = lane & 15;   // M row (A) / N col (B/C/D)
    const int laneHi   = lane >> 4;   // K sub-group select
    const int tokBlock = blockIdx.x * BM;

    if (tid < NEXP) { sMe[tid] = 0.f; sCe[tid] = 0.f; }
    if (tid == 0)   sZ = 0.f;

    v8f acc[4];
#pragma unroll
    for (int n = 0; n < 4; ++n)
#pragma unroll
        for (int j = 0; j < 8; ++j) acc[n][j] = 0.f;

    const int rowA = wave * 16 + laneLow;

#if USE_TDM
    // ---- async pipeline: TDM double-buffer, wave 0 drives the DMA ----
    if (wave == 0) {
#pragma unroll
        for (int p = 0; p < 2; ++p) { // prologue: chunks 0 and 1
            tdm_load_2d((unsigned)(uintptr_t)&sXF[p][0],
                        x + (size_t)tokBlock * DIM + p * KC,
                        2u, DIM, TOKENS, KC, BM, DIM, 5u, 4u);   // fp32, pad 64dw rows by 4dw
            tdm_load_2d((unsigned)(uintptr_t)&sWB[p][0],
                        wbf + (size_t)p * KC,
                        1u, DIM, NEXP, KC, NEXP, DIM, 4u, 4u);   // bf16, pad 32dw rows by 4dw
        }
    }
    for (int ic = 0; ic < NCHUNK; ++ic) {
        if (wave == 0) {
            if (ic + 1 < NCHUNK) __builtin_amdgcn_s_wait_tensorcnt(2); // chunk ic landed
            else                 __builtin_amdgcn_s_wait_tensorcnt(0);
        }
        __syncthreads();
        const int buf = ic & 1;
        const float*          Ab = &sXF[buf][0];
        const unsigned short* Bb = &sWB[buf][0];
#pragma unroll
        for (int kk = 0; kk < KC; kk += 32) {
            const v16bf a = cvt_frag(Ab + rowA * LDAF + kk + laneHi * 8,
                                     Ab + rowA * LDAF + kk + 16 + laneHi * 8);
#pragma unroll
            for (int n = 0; n < 4; ++n) {
                const int e = n * 16 + laneLow;
                const v16bf b = load_frag(Bb + e * LDB + kk + laneHi * 8,
                                          Bb + e * LDB + kk + 16 + laneHi * 8);
                acc[n] = __builtin_amdgcn_wmma_f32_16x16x32_bf16(
                             false, a, false, b, (short)0, acc[n], false, false);
            }
        }
        __syncthreads(); // everyone done reading buf before DMA overwrites it
        if (wave == 0 && ic + 2 < NCHUNK) {
            const int kc = (ic + 2) * KC;
            tdm_load_2d((unsigned)(uintptr_t)&sXF[buf][0],
                        x + (size_t)tokBlock * DIM + kc,
                        2u, DIM, TOKENS, KC, BM, DIM, 5u, 4u);
            tdm_load_2d((unsigned)(uintptr_t)&sWB[buf][0],
                        wbf + (size_t)kc,
                        1u, DIM, NEXP, KC, NEXP, DIM, 4u, 4u);
        }
    }
#else
    // ---- fallback: synchronous cooperative staging ----
    for (int kc = 0; kc < DIM; kc += KC) {
        __syncthreads();
        for (int i = tid; i < BM * (KC / 4); i += NTHREADS) {
            const int r = i / (KC / 4), c = i % (KC / 4);
            reinterpret_cast<float4*>(&sXF[0][r * LDAF])[c] =
                reinterpret_cast<const float4*>(x + (size_t)(tokBlock + r) * DIM + kc)[c];
        }
        for (int i = tid; i < NEXP * (KC / 8); i += NTHREADS) {
            const int r = i / (KC / 8), c = i % (KC / 8);
            reinterpret_cast<uint4*>(&sWB[0][r * LDB])[c] =
                reinterpret_cast<const uint4*>(wbf + (size_t)r * DIM + kc)[c];
        }
        __syncthreads();
#pragma unroll
        for (int kk = 0; kk < KC; kk += 32) {
            const v16bf a = cvt_frag(&sXF[0][rowA * LDAF + kk + laneHi * 8],
                                     &sXF[0][rowA * LDAF + kk + 16 + laneHi * 8]);
#pragma unroll
            for (int n = 0; n < 4; ++n) {
                const int e = n * 16 + laneLow;
                const v16bf b = load_frag(&sWB[0][e * LDB + kk + laneHi * 8],
                                          &sWB[0][e * LDB + kk + 16 + laneHi * 8]);
                acc[n] = __builtin_amdgcn_wmma_f32_16x16x32_bf16(
                             false, a, false, b, (short)0, acc[n], false, false);
            }
        }
    }
#endif

    // ---- epilogue: per-token softmax / top-2 / losses, all in-register ----
    float zAcc = 0.f;
    float meAcc[4] = {0.f, 0.f, 0.f, 0.f};
    float ceAcc[4] = {0.f, 0.f, 0.f, 0.f};
    float* outIdx = out;                               // topk_experts  [TOKENS,2]
    float* outW   = out + (size_t)TOKENS * TOPK;       // combine_weight[TOKENS,2]
    float* outS   = out + (size_t)2 * TOKENS * TOPK;   // scores        [TOKENS,64]

#pragma unroll
    for (int r = 0; r < 8; ++r) {
        const int tok = tokBlock + wave * 16 + r + 8 * laneHi;
        float v[4];
#pragma unroll
        for (int n = 0; n < 4; ++n) v[n] = acc[n][r];

        float m = fmaxf(fmaxf(v[0], v[1]), fmaxf(v[2], v[3]));
#pragma unroll
        for (int off = 1; off < 16; off <<= 1) m = fmaxf(m, __shfl_xor(m, off));

        float e[4], s = 0.f;
#pragma unroll
        for (int n = 0; n < 4; ++n) { e[n] = __expf(v[n] - m); s += e[n]; }
#pragma unroll
        for (int off = 1; off < 16; off <<= 1) s += __shfl_xor(s, off);
        const float inv = 1.0f / s;

        float sc[4];
#pragma unroll
        for (int n = 0; n < 4; ++n) {
            sc[n] = e[n] * inv;
            outS[(size_t)tok * NEXP + n * 16 + laneLow] = sc[n];
            meAcc[n] += sc[n];
            ceAcc[n] += (sc[n] > 0.f) ? 1.f : 0.f;
        }
        if (laneLow == 0) {
            const float lse = m + __logf(s);
            zAcc += lse * lse;
        }

        // top-2 with lower-index tie-break
        float t0 = sc[0], t1 = -INFINITY;
        int   j0 = laneLow, j1 = -1;
#pragma unroll
        for (int n = 1; n < 4; ++n) {
            const float val = sc[n]; const int idx = n * 16 + laneLow;
            if (val > t0 || (val == t0 && idx < j0)) { t1 = t0; j1 = j0; t0 = val; j0 = idx; }
            else if (val > t1 || (val == t1 && idx < j1)) { t1 = val; j1 = idx; }
        }
#pragma unroll
        for (int off = 1; off < 16; off <<= 1) {
            const float o0 = __shfl_xor(t0, off), o1 = __shfl_xor(t1, off);
            const int   q0 = __shfl_xor(j0, off), q1 = __shfl_xor(j1, off);
            if (o0 > t0 || (o0 == t0 && q0 < j0)) { t1 = t0; j1 = j0; t0 = o0; j0 = q0; }
            else if (o0 > t1 || (o0 == t1 && q0 < j1)) { t1 = o0; j1 = q0; }
            if (o1 > t1 || (o1 == t1 && q1 < j1)) { t1 = o1; j1 = q1; }
        }
        if (laneLow == 0) {
            outIdx[(size_t)tok * 2 + 0] = (float)j0;
            outIdx[(size_t)tok * 2 + 1] = (float)j1;
            outW  [(size_t)tok * 2 + 0] = t0;
            outW  [(size_t)tok * 2 + 1] = t1;
        }
    }

    // fold the two half-wave token groups (same experts), then LDS float atomics
#pragma unroll
    for (int n = 0; n < 4; ++n) {
        meAcc[n] += __shfl_xor(meAcc[n], 16);
        ceAcc[n] += __shfl_xor(ceAcc[n], 16);
    }
    if (laneHi == 0) {
#pragma unroll
        for (int n = 0; n < 4; ++n) {
            atomicAdd(&sMe[n * 16 + laneLow], meAcc[n]);  // ds_add_f32
            atomicAdd(&sCe[n * 16 + laneLow], ceAcc[n]);
        }
    }
    zAcc += __shfl_xor(zAcc, 16);
    if (lane == 0) atomicAdd(&sZ, zAcc);
    __syncthreads();

    if (tid < NEXP) {
        atomicAdd(&ws[tid], sMe[tid]);          // per-expert score sums
        atomicAdd(&ws[NEXP + tid], sCe[tid]);   // per-expert nonzero counts
    }
    if (tid == 0) atomicAdd(&ws[2 * NEXP], sZ); // sum of lse^2
}

__global__ void router_init(float* __restrict__ ws) {
    const int i = blockIdx.x * blockDim.x + threadIdx.x;
    if (i < 2 * NEXP + 1) ws[i] = 0.f;
}

__global__ __launch_bounds__(64)
void router_finalize(const float* __restrict__ ws, float* __restrict__ out) {
    __shared__ float red[NEXP];
    const int t = threadIdx.x;
    const float me = ws[t]        * (1.0f / (float)TOKENS);
    const float ce = ws[NEXP + t] * (1.0f / (float)TOKENS);
    red[t] = me * ce;
    __syncthreads();
    for (int s = 32; s > 0; s >>= 1) {
        if (t < s) red[t] += red[t + s];
        __syncthreads();
    }
    if (t == 0) {
        const size_t base = (size_t)2 * TOKENS * TOPK + (size_t)TOKENS * NEXP;
        out[base + 0] = red[0] * (float)NEXP;                  // load_balance_loss
        out[base + 1] = ws[2 * NEXP] * (1.0f / (float)TOKENS); // z_loss
    }
}

extern "C" void kernel_launch(void* const* d_in, const int* in_sizes, int n_in,
                              void* d_out, int out_size, void* d_ws, size_t ws_size,
                              hipStream_t stream) {
    (void)in_sizes; (void)n_in; (void)out_size; (void)ws_size;
    const float* x = (const float*)d_in[0];
    const float* w = (const float*)d_in[1];
    float* out = (float*)d_out;
    float* ws  = (float*)d_ws;
    // ws layout: [0..128] float reduction slots; bf16 w copy at byte offset 512
    unsigned short* wbf = (unsigned short*)((char*)d_ws + 512);

    router_init<<<1, 256, 0, stream>>>(ws);
    convert_w<<<(NEXP * DIM / 4 + 255) / 256, 256, 0, stream>>>(w, wbf);
    router_main<<<TOKENS / BM, NTHREADS, 0, stream>>>(x, wbf, out, ws);
    router_finalize<<<1, 64, 0, stream>>>(ws, out);
}
